// LinearAttention_61168924229644
// MI455X (gfx1250) — matrix operand order repaired and checked by reading, not compiled, not run on hardware
//
#include <hip/hip_runtime.h>
#include <hip/hip_bf16.h>

// ---------------------------------------------------------------------------
// LinearAttention for MI455X (gfx1250, wave32, WMMA 16x16x32 f16)
//   B=32, H=W=64 -> N=4096, C=128, HEADS=4, DH=32, TOKENS = 131072
// ---------------------------------------------------------------------------

typedef _Float16 h16;
typedef __attribute__((ext_vector_type(16))) _Float16 v16h;
typedef __attribute__((ext_vector_type(8)))  float    v8f;

#define TOKENS   131072      // B * H * W
#define NTOK     4096        // H * W per batch
#define CIN      128
#define QKVF     384
#define HEADS    4
#define DH       32
#define LN_EPS   1e-5f
#define INV_SQRT_DH 0.17677669529663689f  // 1/sqrt(32)

static __device__ __forceinline__ v8f wmma16(v16h a, v16h b, v8f c) {
  // D = A(16x32 f16) * B(32x16 f16) + C(16x16 f32)
  return __builtin_amdgcn_wmma_f32_16x16x32_f16(
      /*neg_a=*/false, a, /*neg_b=*/false, b,
      /*c_mod=*/(short)0, c, /*reuse_a=*/false, /*reuse_b=*/false);
}

// A-fragment K index for element e (0..15): K = abase + e + (e>=8 ? 8 : 0)
static __device__ __forceinline__ int akix(int e, int abase) {
  return abase + e + ((e & 8) ? 8 : 0);
}

// ---------------------------------------------------------------------------
// Kernel 0: f32 -> f16 weight conversion
// ---------------------------------------------------------------------------
__global__ __launch_bounds__(256) void cvt_f16_kernel(const float* __restrict__ in,
                                                      h16* __restrict__ out, int n) {
  int i = blockIdx.x * 256 + threadIdx.x;
  if (i < n) out[i] = (h16)in[i];
}

// ---------------------------------------------------------------------------
// Kernel 1: qkv = x @ w_qkv   ([TOKENS x 128] * [128 x 384] -> f16 scratch)
// block = 128 threads (4 waves); each wave computes a 16-token x 384 strip.
// ---------------------------------------------------------------------------
__global__ __launch_bounds__(128) void qkv_gemm_kernel(const float* __restrict__ x,
                                                       const h16* __restrict__ wq,
                                                       h16* __restrict__ qkv) {
  const int lane  = threadIdx.x & 31;
  const int wv    = threadIdx.x >> 5;
  const long tok  = (long)blockIdx.x * 64 + (long)wv * 16;
  const int mrow  = lane & 15;
  const int abase = (lane >> 4) * 8;   // also the D-fragment M offset

  // Load A fragments: x row (f32) -> 4 K-chunks of f16
  const float* xrow = x + (tok + mrow) * CIN;
  v16h a[4];
#pragma unroll
  for (int kk = 0; kk < 4; ++kk) {
#pragma unroll
    for (int e = 0; e < 16; ++e)
      a[kk][e] = (h16)xrow[kk * 32 + akix(e, abase)];
  }

  // 24 output tiles of 16 features each
  for (int ft = 0; ft < QKVF / 16; ++ft) {
    v8f c = {};
#pragma unroll
    for (int kk = 0; kk < 4; ++kk) {
      const h16* brow = wq + (long)(kk * 32) * QKVF + ft * 16 + mrow;
      v16h b;
#pragma unroll
      for (int e = 0; e < 16; ++e) b[e] = brow[(long)akix(e, abase) * QKVF];
      c = wmma16(a[kk], b, c);
    }
#pragma unroll
    for (int r = 0; r < 8; ++r)
      qkv[(tok + r + abase) * QKVF + ft * 16 + mrow] = (h16)c[r];
  }
}

// ---------------------------------------------------------------------------
// Kernel 2: context[b,h] = exp(K)^T V / (sumk * N)    (32x32 per (b,h))
// block = 256 threads (8 waves), one block per (b,h). Waves split n; partial
// 16x16 tiles + sumk reduced through LDS atomics.
// ---------------------------------------------------------------------------
__global__ __launch_bounds__(256) void context_kernel(const h16* __restrict__ qkv,
                                                      h16* __restrict__ ctx) {
  __shared__ float s_ctx[32 * 32];
  __shared__ float s_sumk[32];

  const int bh = blockIdx.x;
  const int b  = bh >> 2;
  const int h  = bh & 3;
  const int tid = threadIdx.x;

  for (int i = tid; i < 1024; i += 256) s_ctx[i] = 0.0f;
  if (tid < 32) s_sumk[tid] = 0.0f;
  __syncthreads();

  const int lane  = tid & 31;
  const int wv    = tid >> 5;
  const int d0    = lane & 15;
  const int abase = (lane >> 4) * 8;

  const h16* kb = qkv + (long)b * NTOK * QKVF + CIN + h * DH;        // k region
  const h16* vb = qkv + (long)b * NTOK * QKVF + 2 * CIN + h * DH;    // v region

  v8f c00 = {}, c01 = {}, c10 = {}, c11 = {};
  float s0 = 0.0f, s1 = 0.0f;

  for (int ch = wv; ch < NTOK / 32; ch += 8) {
    const int nb = ch * 32;
    // A fragments: exp(k) for d-tile0 (d=0..15) and d-tile1 (d=16..31)
    v16h a0, a1;
#pragma unroll
    for (int e = 0; e < 16; ++e) {
      const long n = nb + akix(e, abase);
      const float e0 = __expf((float)kb[n * QKVF + d0]);
      const float e1 = __expf((float)kb[n * QKVF + d0 + 16]);
      s0 += e0; s1 += e1;
      a0[e] = (h16)e0; a1[e] = (h16)e1;
    }
    // B fragments: v (lane holds column e = lane&15, k = n in akix order), two e-tiles
    const h16* vr = vb + (long)(nb + lane) * QKVF;
    __builtin_prefetch(vr + (long)32 * QKVF, 0, 1);
    v16h b0, b1;
#pragma unroll
    for (int e = 0; e < 16; ++e) { b0[e] = vb[(long)(nb + akix(e, abase)) * QKVF + d0]; b1[e] = vb[(long)(nb + akix(e, abase)) * QKVF + 16 + d0]; }

    c00 = wmma16(a0, b0, c00);
    c01 = wmma16(a0, b1, c01);
    c10 = wmma16(a1, b0, c10);
    c11 = wmma16(a1, b1, c11);
  }

  // lanes t and t+16 covered complementary n for the same d -> pairwise reduce
  s0 += __shfl_xor(s0, 16, 32);
  s1 += __shfl_xor(s1, 16, 32);

  // accumulate D fragments into LDS: D[M][N], M = r + abase, N = lane&15
  const int nn = lane & 15;
#pragma unroll
  for (int r = 0; r < 8; ++r) {
    atomicAdd(&s_ctx[(r + abase) * 32 + nn],            c00[r]);
    atomicAdd(&s_ctx[(r + abase) * 32 + 16 + nn],       c01[r]);
    atomicAdd(&s_ctx[(16 + r + abase) * 32 + nn],       c10[r]);
    atomicAdd(&s_ctx[(16 + r + abase) * 32 + 16 + nn],  c11[r]);
  }
  if (lane < 16) {
    atomicAdd(&s_sumk[d0],      s0);
    atomicAdd(&s_sumk[d0 + 16], s1);
  }
  __syncthreads();

  // normalize: context[d][e] /= sumk[d] * N   (folds softmax norm + v/N scale)
  h16* cout = ctx + (long)bh * 1024;
  for (int i = tid; i < 1024; i += 256) {
    const int d = i >> 5;
    cout[i] = (h16)(s_ctx[i] / (s_sumk[d] * (float)NTOK));
  }
}

// ---------------------------------------------------------------------------
// Kernel 3: out = LN( (softmax_d(q)/sqrt(DH)) @ context @ w_out + b_out )
// block = 128 threads (4 waves); each wave owns 16 tokens + 16x128 f32 LDS.
// ---------------------------------------------------------------------------
__global__ __launch_bounds__(128) void out_kernel(const h16* __restrict__ qkv,
                                                  const h16* __restrict__ ctx,
                                                  const h16* __restrict__ wo,
                                                  const float* __restrict__ bout,
                                                  const float* __restrict__ lns,
                                                  float* __restrict__ out) {
  __shared__ float buf[4][16 * CIN];   // 32 KB

  const int lane  = threadIdx.x & 31;
  const int wv    = threadIdx.x >> 5;
  const long tok  = (long)blockIdx.x * 64 + (long)wv * 16;
  const long bat  = tok >> 12;                 // token / 4096
  const int mrow  = lane & 15;
  const int abase = (lane >> 4) * 8;
  float* mybuf = buf[wv];

  // ---- phase 1: out_pre[token, h*32+e] = softmax(q) . context_h -------------
  const h16* qrow = qkv + (tok + mrow) * QKVF;   // q region = features [0,128)
  v8f acc[8];
#pragma unroll
  for (int h = 0; h < HEADS; ++h) {
    float ev[16];
    float s = 0.0f;
#pragma unroll
    for (int e = 0; e < 16; ++e) {
      const float v = __expf((float)qrow[h * DH + akix(e, abase)]);
      ev[e] = v; s += v;
    }
    s += __shfl_xor(s, 16, 32);                 // lanes t / t+16: same token
    const float inv = INV_SQRT_DH / s;
    v16h a;
#pragma unroll
    for (int e = 0; e < 16; ++e) a[e] = (h16)(ev[e] * inv);

    const h16* chh = ctx + (bat * HEADS + h) * 1024;  // [32 d][32 e] row-major
    v16h b0, b1;
#pragma unroll
    for (int e = 0; e < 16; ++e) { b0[e] = chh[akix(e, abase) * 32 + mrow]; b1[e] = chh[akix(e, abase) * 32 + 16 + mrow]; }
    v8f z0 = {}, z1 = {};
    acc[h * 2]     = wmma16(a, b0, z0);
    acc[h * 2 + 1] = wmma16(a, b1, z1);
  }
  // spill out_pre to LDS row-major [token16][128] for layout conversion D->A
#pragma unroll
  for (int t = 0; t < 8; ++t)
#pragma unroll
    for (int r = 0; r < 8; ++r)
      mybuf[(r + abase) * CIN + t * 16 + mrow] = acc[t][r];
  __syncthreads();

  // ---- phase 2: out = out_pre @ w_out + b_out -------------------------------
  v16h a2[4];
#pragma unroll
  for (int kk = 0; kk < 4; ++kk)
#pragma unroll
    for (int e = 0; e < 16; ++e)
      a2[kk][e] = (h16)mybuf[mrow * CIN + kk * 32 + akix(e, abase)];

#pragma unroll
  for (int f = 0; f < 8; ++f) {
    v8f c = {};
#pragma unroll
    for (int kk = 0; kk < 4; ++kk) {
      const h16* brow = wo + (long)(kk * 32) * CIN + f * 16 + mrow;
      v16h b;
#pragma unroll
      for (int e = 0; e < 16; ++e) b[e] = brow[(long)akix(e, abase) * CIN];
      c = wmma16(a2[kk], b, c);
    }
    const float bias = bout[f * 16 + mrow];
#pragma unroll
    for (int r = 0; r < 8; ++r)
      mybuf[(r + abase) * CIN + f * 16 + mrow] = c[r] + bias;
  }
  __syncthreads();

  // ---- phase 3: LayerNorm over channels -------------------------------------
  const int t   = lane & 15;       // token within wave tile
  const int hs  = lane >> 4;       // which 64-channel half
  const float* row = mybuf + t * CIN;
  float s = 0.0f, ss = 0.0f;
#pragma unroll 8
  for (int i = 0; i < 64; ++i) {
    const float v = row[hs * 64 + i];
    s += v; ss += v * v;
  }
  s  += __shfl_xor(s, 16, 32);
  ss += __shfl_xor(ss, 16, 32);
  const float mean = s * (1.0f / CIN);
  const float var  = ss * (1.0f / CIN) - mean * mean;
  const float rstd = rsqrtf(var + LN_EPS);
  float* orow = out + (tok + t) * CIN;
#pragma unroll 8
  for (int i = 0; i < 64; ++i) {
    const int cix = hs * 64 + i;
    orow[cix] = (row[cix] - mean) * rstd * lns[cix];
  }
}

// ---------------------------------------------------------------------------
// launcher
// ---------------------------------------------------------------------------
extern "C" void kernel_launch(void* const* d_in, const int* in_sizes, int n_in,
                              void* d_out, int out_size, void* d_ws, size_t ws_size,
                              hipStream_t stream) {
  (void)in_sizes; (void)n_in; (void)out_size; (void)ws_size;
  const float* x        = (const float*)d_in[0];   // [32,64,64,128]
  const float* w_qkv    = (const float*)d_in[1];   // [128,384]
  const float* w_out    = (const float*)d_in[2];   // [128,128]
  const float* b_out    = (const float*)d_in[3];   // [128]
  const float* ln_scale = (const float*)d_in[4];   // [128]
  float* out = (float*)d_out;

  // workspace layout (f16 scratch; qkv tensor fits in MI455X's 192MB L2)
  h16* qkv_h = (h16*)d_ws;                                    // 131072*384 halves
  h16* wq_h  = (h16*)((char*)d_ws + (size_t)TOKENS * QKVF * 2);
  h16* wo_h  = wq_h + (size_t)CIN * QKVF;
  h16* ctx_h = wo_h + (size_t)CIN * CIN;                      // 32*4*32*32 halves

  cvt_f16_kernel<<<(CIN * QKVF + 255) / 256, 256, 0, stream>>>(w_qkv, wq_h, CIN * QKVF);
  cvt_f16_kernel<<<(CIN * CIN + 255) / 256, 256, 0, stream>>>(w_out, wo_h, CIN * CIN);

  qkv_gemm_kernel<<<TOKENS / 64, 128, 0, stream>>>(x, wq_h, qkv_h);
  context_kernel<<<32 * HEADS, 256, 0, stream>>>(qkv_h, ctx_h);
  out_kernel<<<TOKENS / 64, 128, 0, stream>>>(qkv_h, ctx_h, wo_h, b_out, ln_scale, out);
}
